// Attention_33938831573672
// MI455X (gfx1250) — matrix operand order repaired and checked by reading, not compiled, not run on hardware
//
#include <hip/hip_runtime.h>
#include <math.h>

#define B_   4
#define S_   4096
#define DIM_ 768
#define H_   12
#define D_   64
#define NH_  2
#define L_   64
#define NB_  128
#define T_   (NH_*S_)     /* 8192 */
#define C_   (T_/L_)      /* 128  */
#define BH_  (B_*H_)      /* 48   */
#define M_   (B_*S_)      /* 16384*/

#define MASK_VAL  (-1e9f)
#define SELF_MASK (-1e5f)

typedef __bf16 bf16;
typedef __attribute__((ext_vector_type(16))) __bf16 v16bf;
typedef __attribute__((ext_vector_type(8)))  __bf16 v8bf;
typedef __attribute__((ext_vector_type(8)))  float  v8f;

__device__ __forceinline__ v8f wmma_bf16(v16bf a, v16bf b, v8f c) {
  // D = A(16x32) * B(32x16) + C(16x16), f32 accum
  return __builtin_amdgcn_wmma_f32_16x16x32_bf16(false, a, false, b, (short)0, c,
                                                 false, false);
}

// A fragment (16x32 bf16, MxK): lane<16 -> M=lane, K={0..7,16..23};
// lane>=16 -> M=lane-16, K={8..15,24..31}. base points at (row0,k0), ld = row stride.
__device__ __forceinline__ v16bf load_frag_a(const bf16* base, int ld, int lane) {
  const int half = lane >> 4, m = lane & 15;
  const bf16* p = base + m * ld + half * 8;
  v8bf lo = *(const v8bf*)(p);
  v8bf hi = *(const v8bf*)(p + 16);
  v16bf a;
#pragma unroll
  for (int i = 0; i < 8; ++i) { a[i] = lo[i]; a[i + 8] = hi[i]; }
  return a;
}

// B fragment (32x16 bf16, KxN) from "Bt" storage where row n holds column n of B
// (i.e. Bt[n][k] = B[k][n]). lane<16 -> N=lane, K={0..7,16..23}; lane>=16 -> N=lane-16, K={8..15,24..31}.
__device__ __forceinline__ v16bf load_frag_b(const bf16* baseT, int ld, int lane) {
  const int half = lane >> 4, n = lane & 15;
  const bf16* p = baseT + n * ld + half * 8;
  v8bf lo = *(const v8bf*)(p);
  v8bf hi = *(const v8bf*)(p + 16);
  v16bf b;
#pragma unroll
  for (int i = 0; i < 8; ++i) { b[i] = lo[i]; b[i + 8] = hi[i]; }
  return b;
}

// ---------------------------------------------------------------------------
// Kernel 1: qk = X @ Wq^T, v = X @ Wv^T   (both [B,H,S,D] f32 out, bf16 WMMA)
// grid (M/64, H), block 256. Waves 0-3 -> qk strips, waves 4-7 -> v strips.
// ---------------------------------------------------------------------------
__global__ __launch_bounds__(256) void k_proj_qkv(
    const float* __restrict__ X, const float* __restrict__ Wq,
    const float* __restrict__ Wv, float* __restrict__ qk, float* __restrict__ v) {
  __shared__ __align__(16) bf16 Xs[64][64];
  __shared__ __align__(16) bf16 Wqs[64][64];
  __shared__ __align__(16) bf16 Wvs[64][64];

  const int tid = threadIdx.x, lane = tid & 31, w = tid >> 5;
  const int m0 = blockIdx.x * 64;
  const int h = blockIdx.y;
  const int tgt = w >> 2;
  const int mstrip = (w & 3) * 16;

  const v8f vz = {0.f, 0.f, 0.f, 0.f, 0.f, 0.f, 0.f, 0.f};
  v8f acc[4];
#pragma unroll
  for (int i = 0; i < 4; ++i) acc[i] = vz;

  for (int kt = 0; kt < 12; ++kt) {
    const int k0 = kt * 64;
#pragma unroll
    for (int i = 0; i < 4; ++i) {
      int lin = tid + 256 * i;  // 0..1023
      int row = lin >> 4, c4 = (lin & 15) * 4;
      float4 xv = *(const float4*)(X + (size_t)(m0 + row) * DIM_ + k0 + c4);
      Xs[row][c4 + 0] = (bf16)xv.x; Xs[row][c4 + 1] = (bf16)xv.y;
      Xs[row][c4 + 2] = (bf16)xv.z; Xs[row][c4 + 3] = (bf16)xv.w;
      float4 qv = *(const float4*)(Wq + (size_t)(h * 64 + row) * DIM_ + k0 + c4);
      Wqs[row][c4 + 0] = (bf16)qv.x; Wqs[row][c4 + 1] = (bf16)qv.y;
      Wqs[row][c4 + 2] = (bf16)qv.z; Wqs[row][c4 + 3] = (bf16)qv.w;
      float4 wv = *(const float4*)(Wv + (size_t)(h * 64 + row) * DIM_ + k0 + c4);
      Wvs[row][c4 + 0] = (bf16)wv.x; Wvs[row][c4 + 1] = (bf16)wv.y;
      Wvs[row][c4 + 2] = (bf16)wv.z; Wvs[row][c4 + 3] = (bf16)wv.w;
    }
    if (kt + 1 < 12) {  // warm L2/WGP$ for the next K tile (global_prefetch_b8)
      __builtin_prefetch(X + (size_t)(m0 + (tid >> 2)) * DIM_ + k0 + 64, 0, 1);
      __builtin_prefetch(Wq + (size_t)(h * 64 + (tid >> 2)) * DIM_ + k0 + 64, 0, 1);
    }
    __syncthreads();
    const bf16* Ws = tgt ? &Wvs[0][0] : &Wqs[0][0];
    v16bf a0 = load_frag_a(&Xs[mstrip][0], 64, lane);
    v16bf a1 = load_frag_a(&Xs[mstrip][32], 64, lane);
#pragma unroll
    for (int nt = 0; nt < 4; ++nt) {
      v16bf b0 = load_frag_b(Ws + nt * 16 * 64, 64, lane);
      v16bf b1 = load_frag_b(Ws + nt * 16 * 64 + 32, 64, lane);
      acc[nt] = wmma_bf16(a0, b0, acc[nt]);
      acc[nt] = wmma_bf16(a1, b1, acc[nt]);
    }
    __syncthreads();
  }

  const int bb = m0 >> 12;          // m0 / S
  const int sbase = m0 & (S_ - 1);
  float* outp = (tgt ? v : qk) + (size_t)(bb * H_ + h) * S_ * D_;
  const int halfId = lane >> 4, nloc = lane & 15;
#pragma unroll
  for (int nt = 0; nt < 4; ++nt) {
    const int d = nt * 16 + nloc;
#pragma unroll
    for (int r = 0; r < 8; ++r) {
      const int Mrow = mstrip + halfId * 8 + r;
      outp[(size_t)(sbase + Mrow) * D_ + d] = acc[nt][r];
    }
  }
}

// ---------------------------------------------------------------------------
// Kernel 2: LSH bucketing. One wave per (b,h,nh,s). 8 waves/block.
// ---------------------------------------------------------------------------
__global__ __launch_bounds__(256) void k_hash(
    const float* __restrict__ qk, const float* __restrict__ rot,
    int* __restrict__ buckets) {
  const int lane = threadIdx.x & 31;
  const int wv = threadIdx.x >> 5;
  const int idx = blockIdx.x * 8 + wv;        // ((b*H+h)*NH + nh)*S + s
  const int s = idx & (S_ - 1);
  const int nh = (idx >> 12) & 1;
  const int hh = (idx >> 13) % H_;
  const int b = idx / (S_ * NH_ * H_);

  const float* qrow = qk + ((size_t)(b * H_ + hh) * S_ + s) * D_;
  const float q0 = qrow[lane], q1 = qrow[lane + 32];
  const float* rbase = rot + ((size_t)hh * D_ * NH_ + nh) * 64;  // stride per d: NH*64

  float a0 = 0.f, a1 = 0.f;
#pragma unroll 4
  for (int d = 0; d < 64; ++d) {
    float qv = __shfl((d < 32) ? q0 : q1, d & 31, 32);
    const float* rp = rbase + (size_t)d * (NH_ * 64);
    a0 = fmaf(qv, rp[lane], a0);
    a1 = fmaf(qv, rp[lane + 32], a1);
  }
  // argmax over concat([r, -r]) with first-index tie-break
  float bv = a0; int bi = lane;
  if (a1 > bv)  { bv = a1;  bi = lane + 32; }
  if (-a0 > bv) { bv = -a0; bi = lane + 64; }
  if (-a1 > bv) { bv = -a1; bi = lane + 96; }
#pragma unroll
  for (int o = 16; o >= 1; o >>= 1) {
    float ov = __shfl_xor(bv, o, 32);
    int oi = __shfl_xor(bi, o, 32);
    if (ov > bv || (ov == bv && oi < bi)) { bv = ov; bi = oi; }
  }
  if (lane == 0)
    buckets[(size_t)(b * H_ + hh) * T_ + nh * S_ + s] = bi + nh * NB_;
}

// ---------------------------------------------------------------------------
// Kernel 3: stable counting sort per (b,h). Keys = bucket (256 bins); within a
// bucket items are already in ascending t order -> chunked LDS histogram sort.
// ---------------------------------------------------------------------------
#define SCH 32
#define SIC (T_ / SCH)  /* 256 */
__global__ __launch_bounds__(256) void k_sort(
    const int* __restrict__ buckets, int* __restrict__ st, int* __restrict__ undo) {
  __shared__ int hist[SCH][256];
  __shared__ int binOff[256];
  const int tid = threadIdx.x;
  const int bh = blockIdx.x;
  const int* bk = buckets + (size_t)bh * T_;

  for (int i = tid; i < SCH * 256; i += 256) ((int*)hist)[i] = 0;
  __syncthreads();
  if (tid < SCH) {
    const int base = tid * SIC;
    for (int i = 0; i < SIC; ++i) hist[tid][bk[base + i]]++;
  }
  __syncthreads();
  {  // per-bin exclusive scan over chunks; bin totals -> binOff
    int tot = 0;
    for (int c = 0; c < SCH; ++c) { int x = hist[c][tid]; hist[c][tid] = tot; tot += x; }
    binOff[tid] = tot;
  }
  __syncthreads();
  if (tid == 0) {
    int acc = 0;
    for (int i = 0; i < 256; ++i) { int x = binOff[i]; binOff[i] = acc; acc += x; }
  }
  __syncthreads();
  if (tid < SCH) {
    const int base = tid * SIC;
    int* ho = hist[tid];
    int* stp = st + (size_t)bh * T_;
    int* unp = undo + (size_t)bh * T_;
    for (int i = 0; i < SIC; ++i) {
      const int t = base + i;
      const int bkt = bk[t];
      const int pos = binOff[bkt] + ho[bkt]++;
      stp[pos] = t & (S_ - 1);
      unp[t] = pos;
    }
  }
}

// ---------------------------------------------------------------------------
// Kernel 4: chunked attention. One block (4 waves) per (b,h,chunk).
// Gather+normalize -> LDS bf16, QK^T WMMA, mask, logsumexp, probs -> LDS,
// PV WMMA, store o & logits in sorted order.
// ---------------------------------------------------------------------------
__global__ __launch_bounds__(128) void k_attn(
    const float* __restrict__ qk, const float* __restrict__ vv,
    const float* __restrict__ mask, const int* __restrict__ st,
    float* __restrict__ o_g, float* __restrict__ logit_g) {
  __shared__ __align__(16) bf16 qs[64][64];    // q rows (A layout)
  __shared__ __align__(16) bf16 ks[128][64];   // k rows (Bt layout for scores)
  __shared__ __align__(16) bf16 vt[64][128];   // v transposed (Bt layout for PV)
  __shared__ __align__(16) bf16 ps[64][128];   // probs (A layout for PV)
  __shared__ int qi_s[64];
  __shared__ int ki_s[128];
  __shared__ int km_s[128];

  const int tid = threadIdx.x, lane = tid & 31, w = tid >> 5;
  const int bh = blockIdx.x / C_;
  const int c = blockIdx.x % C_;
  const int b = bh / H_;
  const int cp = (c + C_ - 1) % C_;
  const int* stp = st + (size_t)bh * T_;
  const float* qbase = qk + (size_t)bh * S_ * D_;
  const float* vbase = vv + (size_t)bh * S_ * D_;

  {  // gather k/v rows: thread tid -> key row m (prev chunk first, then current)
    const int m = tid;
    const int tk = (m < L_) ? (cp * L_ + m) : (c * L_ + (m - L_));
    const int sk = stp[tk];
    const float* qr = qbase + (size_t)sk * D_;
    float buf[64];
    float ss = 0.f;
#pragma unroll
    for (int i = 0; i < 16; ++i) {
      float4 x = *(const float4*)(qr + i * 4);
      buf[i * 4 + 0] = x.x; buf[i * 4 + 1] = x.y;
      buf[i * 4 + 2] = x.z; buf[i * 4 + 3] = x.w;
      ss += x.x * x.x + x.y * x.y + x.z * x.z + x.w * x.w;
    }
    const float scale = rsqrtf(ss * (1.f / 64.f) + 1e-6f) * 0.125f;  // 1/sqrt(D)
#pragma unroll
    for (int d = 0; d < 64; ++d) ks[m][d] = (bf16)(buf[d] * scale);
    const float* vr = vbase + (size_t)sk * D_;
#pragma unroll
    for (int i = 0; i < 16; ++i) {
      float4 x = *(const float4*)(vr + i * 4);
      vt[i * 4 + 0][m] = (bf16)x.x; vt[i * 4 + 1][m] = (bf16)x.y;
      vt[i * 4 + 2][m] = (bf16)x.z; vt[i * 4 + 3][m] = (bf16)x.w;
    }
    ki_s[m] = sk;
    km_s[m] = (mask[(size_t)b * S_ + sk] > 0.5f) ? 1 : 0;
  }
  if (tid < L_) {  // gather q rows
    const int row = tid;
    const int sq = stp[c * L_ + row];
    qi_s[row] = sq;
    const float* qr = qbase + (size_t)sq * D_;
#pragma unroll
    for (int i = 0; i < 16; ++i) {
      float4 x = *(const float4*)(qr + i * 4);
      qs[row][i * 4 + 0] = (bf16)x.x; qs[row][i * 4 + 1] = (bf16)x.y;
      qs[row][i * 4 + 2] = (bf16)x.z; qs[row][i * 4 + 3] = (bf16)x.w;
    }
  }
  __syncthreads();

  const int mstrip = w * 16;
  const int halfId = lane >> 4, nloc = lane & 15;
  const v8f vz = {0.f, 0.f, 0.f, 0.f, 0.f, 0.f, 0.f, 0.f};

  // ---- scores = q @ k^T  (16 x 128 per wave) ----
  v16bf a0 = load_frag_a(&qs[mstrip][0], 64, lane);
  v16bf a1 = load_frag_a(&qs[mstrip][32], 64, lane);
  v8f sc[8];
#pragma unroll
  for (int nt = 0; nt < 8; ++nt) {
    v16bf b0 = load_frag_b(&ks[nt * 16][0], 64, lane);
    v16bf b1 = load_frag_b(&ks[nt * 16][32], 64, lane);
    v8f t0 = wmma_bf16(a0, b0, vz);
    sc[nt] = wmma_bf16(a1, b1, t0);
  }

  // ---- masking + per-row logsumexp + probs ----
#pragma unroll
  for (int r = 0; r < 8; ++r) {
    const int Mrow = mstrip + halfId * 8 + r;
    const int qiv = qi_s[Mrow];
    float vals[8];
    float mx = -3.0e38f;
#pragma unroll
    for (int nt = 0; nt < 8; ++nt) {
      const int N = nt * 16 + nloc;
      float x = sc[nt][r];
      if (!km_s[N]) x = MASK_VAL;
      if (qiv == ki_s[N]) x = SELF_MASK;
      vals[nt] = x;
      mx = fmaxf(mx, x);
    }
#pragma unroll
    for (int o = 8; o >= 1; o >>= 1) mx = fmaxf(mx, __shfl_xor(mx, o, 32));
    float se = 0.f;
#pragma unroll
    for (int nt = 0; nt < 8; ++nt) se += __expf(vals[nt] - mx);
#pragma unroll
    for (int o = 8; o >= 1; o >>= 1) se += __shfl_xor(se, o, 32);
    const float logit = mx + __logf(se);
#pragma unroll
    for (int nt = 0; nt < 8; ++nt)
      ps[Mrow][nt * 16 + nloc] = (bf16)__expf(vals[nt] - logit);
    if (nloc == 0) logit_g[(size_t)bh * T_ + c * L_ + Mrow] = logit;
  }
  __syncthreads();

  // ---- o = probs @ v  (16 x 64 per wave) ----
  v16bf ap[4];
#pragma unroll
  for (int kk = 0; kk < 4; ++kk) ap[kk] = load_frag_a(&ps[mstrip][kk * 32], 128, lane);
#pragma unroll
  for (int nt = 0; nt < 4; ++nt) {
    v8f oc = vz;
#pragma unroll
    for (int kk = 0; kk < 4; ++kk) {
      v16bf bv = load_frag_b(&vt[nt * 16][kk * 32], 128, lane);
      oc = wmma_bf16(ap[kk], bv, oc);
    }
#pragma unroll
    for (int r = 0; r < 8; ++r) {
      const int Mrow = mstrip + halfId * 8 + r;
      const int d = nt * 16 + nloc;
      o_g[((size_t)bh * T_ + c * L_ + Mrow) * D_ + d] = oc[r];
    }
  }
}

// ---------------------------------------------------------------------------
// Kernel 5: undo sort + softmax-weighted combine of hash rounds.
// ---------------------------------------------------------------------------
__global__ __launch_bounds__(256) void k_combine(
    const float* __restrict__ o_g, const float* __restrict__ logit_g,
    const int* __restrict__ undo, float* __restrict__ oc) {
  const int gid = blockIdx.x * 256 + threadIdx.x;  // BH*S
  const int bh = gid >> 12;
  const int s = gid & (S_ - 1);
  const int b = bh / H_, h = bh % H_;
  const int* un = undo + (size_t)bh * T_;
  const int p0 = un[s], p1 = un[S_ + s];
  const float* lg = logit_g + (size_t)bh * T_;
  const float l0 = lg[p0], l1 = lg[p1];
  const float mx = fmaxf(l0, l1);
  const float e0 = __expf(l0 - mx), e1 = __expf(l1 - mx);
  const float inv = 1.f / (e0 + e1);
  const float w0 = e0 * inv, w1 = e1 * inv;
  const float4* a = (const float4*)(o_g + ((size_t)bh * T_ + p0) * D_);
  const float4* bb = (const float4*)(o_g + ((size_t)bh * T_ + p1) * D_);
  float4* op = (float4*)(oc + (size_t)(b * S_ + s) * DIM_ + h * D_);
#pragma unroll
  for (int i = 0; i < 16; ++i) {
    float4 x = a[i], y = bb[i], z;
    z.x = w0 * x.x + w1 * y.x; z.y = w0 * x.y + w1 * y.y;
    z.z = w0 * x.z + w1 * y.z; z.w = w0 * x.w + w1 * y.w;
    op[i] = z;
  }
}

// ---------------------------------------------------------------------------
// Kernel 6: Y = out_comb @ Wff^T + bff  (bf16 WMMA GEMM, f32 out)
// ---------------------------------------------------------------------------
__global__ __launch_bounds__(256) void k_proj_out(
    const float* __restrict__ A, const float* __restrict__ Wff,
    const float* __restrict__ bff, float* __restrict__ Y) {
  __shared__ __align__(16) bf16 As[64][64];
  __shared__ __align__(16) bf16 Ws[64][64];
  const int tid = threadIdx.x, lane = tid & 31, w = tid >> 5;
  const int m0 = blockIdx.x * 64;
  const int n0 = blockIdx.y * 64;
  const int mstrip = (w & 3) * 16;
  const int npair = (w >> 2) * 2;

  const v8f vz = {0.f, 0.f, 0.f, 0.f, 0.f, 0.f, 0.f, 0.f};
  v8f acc[2];
  acc[0] = vz; acc[1] = vz;

  for (int kt = 0; kt < 12; ++kt) {
    const int k0 = kt * 64;
#pragma unroll
    for (int i = 0; i < 4; ++i) {
      int lin = tid + 256 * i;
      int row = lin >> 4, c4 = (lin & 15) * 4;
      float4 av = *(const float4*)(A + (size_t)(m0 + row) * DIM_ + k0 + c4);
      As[row][c4 + 0] = (bf16)av.x; As[row][c4 + 1] = (bf16)av.y;
      As[row][c4 + 2] = (bf16)av.z; As[row][c4 + 3] = (bf16)av.w;
      float4 wv = *(const float4*)(Wff + (size_t)(n0 + row) * DIM_ + k0 + c4);
      Ws[row][c4 + 0] = (bf16)wv.x; Ws[row][c4 + 1] = (bf16)wv.y;
      Ws[row][c4 + 2] = (bf16)wv.z; Ws[row][c4 + 3] = (bf16)wv.w;
    }
    if (kt + 1 < 12)
      __builtin_prefetch(A + (size_t)(m0 + (tid >> 2)) * DIM_ + k0 + 64, 0, 1);
    __syncthreads();
    v16bf a0 = load_frag_a(&As[mstrip][0], 64, lane);
    v16bf a1 = load_frag_a(&As[mstrip][32], 64, lane);
#pragma unroll
    for (int j = 0; j < 2; ++j) {
      const int nt = npair + j;
      v16bf b0 = load_frag_b(&Ws[nt * 16][0], 64, lane);
      v16bf b1 = load_frag_b(&Ws[nt * 16][32], 64, lane);
      acc[j] = wmma_bf16(a0, b0, acc[j]);
      acc[j] = wmma_bf16(a1, b1, acc[j]);
    }
    __syncthreads();
  }
  const int halfId = lane >> 4, nloc = lane & 15;
#pragma unroll
  for (int j = 0; j < 2; ++j) {
#pragma unroll
    for (int r = 0; r < 8; ++r) {
      const int row = m0 + mstrip + halfId * 8 + r;
      const int col = n0 + (npair + j) * 16 + nloc;
      Y[(size_t)row * DIM_ + col] = acc[j][r] + bff[col];
    }
  }
}

// ---------------------------------------------------------------------------
extern "C" void kernel_launch(void* const* d_in, const int* in_sizes, int n_in,
                              void* d_out, int out_size, void* d_ws, size_t ws_size,
                              hipStream_t stream) {
  (void)in_sizes; (void)n_in; (void)out_size; (void)ws_size;
  const float* X    = (const float*)d_in[0];
  const float* mask = (const float*)d_in[1];
  const float* Wq   = (const float*)d_in[2];
  const float* Wv   = (const float*)d_in[3];
  const float* Wff  = (const float*)d_in[4];
  const float* bff  = (const float*)d_in[5];
  const float* rot  = (const float*)d_in[6];
  float* Y = (float*)d_out;

  char* ws = (char*)d_ws;
  size_t off = 0;
  auto take = [&](size_t bytes) -> char* {
    off = (off + 255) & ~(size_t)255;
    char* p = ws + off;
    off += bytes;
    return p;
  };
  float* qk      = (float*)take((size_t)BH_ * S_ * D_ * sizeof(float));
  float* v       = (float*)take((size_t)BH_ * S_ * D_ * sizeof(float));
  float* ocmb    = (float*)take((size_t)M_ * DIM_ * sizeof(float));
  float* o_g     = (float*)take((size_t)BH_ * T_ * D_ * sizeof(float));
  float* lg      = (float*)take((size_t)BH_ * T_ * sizeof(float));
  int*   buckets = (int*)take((size_t)BH_ * T_ * sizeof(int));
  int*   st      = (int*)take((size_t)BH_ * T_ * sizeof(int));
  int*   undo    = (int*)take((size_t)BH_ * T_ * sizeof(int));

  k_proj_qkv<<<dim3(M_ / 64, H_), 256, 0, stream>>>(X, Wq, Wv, qk, v);
  k_hash<<<dim3((BH_ * NH_ * S_) / 8), 256, 0, stream>>>(qk, rot, buckets);
  k_sort<<<dim3(BH_), 256, 0, stream>>>(buckets, st, undo);
  k_attn<<<dim3(BH_ * C_), 128, 0, stream>>>(qk, v, mask, st, o_g, lg);
  k_combine<<<dim3((BH_ * S_) / 256), 256, 0, stream>>>(o_g, lg, undo, ocmb);
  k_proj_out<<<dim3(M_ / 64, DIM_ / 64), 256, 0, stream>>>(ocmb, Wff, bff, Y);
}
